// SpectralMambaBlock_10960756539674
// MI455X (gfx1250) — compile-verified, matrix operand order checked by reading, not run on hardware
//
#include <hip/hip_runtime.h>
#include <math.h>

// ---------------------------------------------------------------------------
// CDNA5 (gfx1250) wave32 WMMA helpers: D(16x16,f32) = A(16x32,bf16)*B(32x16,bf16)+C
// A frag (ISA 7.12.2): lane half h=lane>>4, row=lane&15; elems 0..7 -> K=h*8+i,
// elems 8..15 -> K=16+h*8+(i-8). B assumed mirrored (lane = column n, same K map).
// C/D: N=lane&15, VGPR r -> M = r + 8*(lane>>4).
// ---------------------------------------------------------------------------
typedef __bf16 bf16;
typedef __attribute__((ext_vector_type(16))) bf16  v16bf;
typedef __attribute__((ext_vector_type(8)))  float v8f;

__device__ __forceinline__ v16bf frag_ld(const bf16* p, int half) {
  v16bf v;
#pragma unroll
  for (int i = 0; i < 8; ++i) v[i] = p[half * 8 + i];
#pragma unroll
  for (int i = 0; i < 8; ++i) v[i + 8] = p[16 + half * 8 + i];
  return v;
}

__device__ __forceinline__ v8f wmma_bf16(v16bf a, v16bf b, v8f c) {
  // (neg_a, A, neg_b, B, c_mod, C, reuse_a, reuse_b)
  return __builtin_amdgcn_wmma_f32_16x16x32_bf16(false, a, false, b, (short)0, c,
                                                 false, false);
}

__device__ __forceinline__ float sigmoidf_(float x) { return 1.0f / (1.0f + __expf(-x)); }
__device__ __forceinline__ float siluf_(float x)    { return x * sigmoidf_(x); }
__device__ __forceinline__ float softplusf_(float x){ return (x > 20.0f) ? x : log1pf(__expf(x)); }
__device__ __forceinline__ float geluf_(float x)    { return 0.5f * x * (1.0f + erff(x * 0.70710678118654752f)); }

// Problem constants
#define NB       2
#define DIMC     64
#define DINNER   128
#define DSTATE   16
#define DTRANK   4
#define HH       32
#define WW       32
#define DDp      32
#define TT       16                  // tokens per WMMA tile
#define CH_STRIDE 32768              // H*W*DD
#define B_STRIDE  2097152            // 64*H*W*DD
#define NPOS     65536               // B*H*W*DD

// ---------------------------------------------------------------------------
// One workgroup = one mamba sequence. 128 threads = 4 wave32.
// mode 0: spatial  (n -> b=n>>5, dd=n&31, token l = hw),  L = 1024
// mode 1: spectral (n -> b=n>>10, hw=n&1023, token l=dd), L = 32
// Element addr of (token l, channel c) in (B,64,H,W,DD): base0 + l*lstride + c*32768
// ---------------------------------------------------------------------------
__global__ __launch_bounds__(128)
void smb_mamba_kernel(const float* __restrict__ x,
                      const float* __restrict__ W_in,   // (256,64)
                      const float* __restrict__ conv_w, // (128,4)
                      const float* __restrict__ conv_b, // (128)
                      const float* __restrict__ W_x,    // (36,128)
                      const float* __restrict__ W_dt,   // (128,4)
                      const float* __restrict__ b_dt,   // (128)
                      const float* __restrict__ A_log,  // (128,16)
                      const float* __restrict__ Dp,     // (128)
                      const float* __restrict__ W_out,  // (64,128)
                      float* __restrict__ out,          // (B,64,H,W,DD)
                      int mode, int L)
{
  __shared__ bf16  sWin [256 * 64];   // W_in  bf16 (row n over k)
  __shared__ bf16  sWx  [48 * 128];   // W_x   bf16, N padded 36->48 with zeros
  __shared__ bf16  sWout[64 * 128];   // W_out bf16
  __shared__ float sA   [128 * 16];   // A = -exp(A_log)
  __shared__ float sWdt [128 * 4];
  __shared__ float sbdt [128];
  __shared__ float scw  [128 * 4];
  __shared__ float scb  [128];
  __shared__ float sDp  [128];
  __shared__ bf16  tA   [16 * 64];    // x tile (tokens x DIM)
  __shared__ float sxz  [16 * 256];   // GEMM1 out: [0:128)=conv input, [128:256)=z
  __shared__ float sxc  [16 * 128];   // silu(conv) f32
  __shared__ bf16  sxcb [16 * 128];   // silu(conv) bf16 (GEMM2 A)
  __shared__ float sdbl [16 * 48];    // dt_raw|B|C (cols 0..35 valid)
  __shared__ float sdt  [16 * 128];   // softplus dt
  __shared__ bf16  syb  [16 * 128];   // gated y bf16 (GEMM3 A)
  __shared__ float shist[3 * 128];    // last 3 raw conv inputs (across tiles)

  const int tid  = threadIdx.x;
  const int lane = tid & 31;
  const int half = (lane >> 4) & 1;
  const int wid  = tid >> 5;
  const int row  = lane & 15;
  const int n    = blockIdx.x;

  // ---- preload weights / derived params into LDS -------------------------
  for (int e = tid; e < 256 * 64; e += 128) sWin[e] = (bf16)W_in[e];
  for (int e = tid; e < 48 * 128; e += 128) {
    int r = e >> 7;
    sWx[e] = (r < 36) ? (bf16)W_x[e] : (bf16)0.0f;
  }
  for (int e = tid; e < 64 * 128; e += 128) sWout[e] = (bf16)W_out[e];
  for (int e = tid; e < 128 * 16; e += 128) sA[e] = -__expf(A_log[e]);
  for (int e = tid; e < 128 * 4;  e += 128) { sWdt[e] = W_dt[e]; scw[e] = conv_w[e]; }
  if (tid < 128) { sbdt[tid] = b_dt[tid]; scb[tid] = conv_b[tid]; sDp[tid] = Dp[tid]; }
  for (int e = tid; e < 3 * 128; e += 128) shist[e] = 0.0f;

  long base0;
  int  lstride;
  if (mode == 0) { base0 = (long)(n >> 5)  * B_STRIDE + (n & 31);            lstride = 32; }
  else           { base0 = (long)(n >> 10) * B_STRIDE + (long)(n & 1023) * 32; lstride = 1; }

  float h[DSTATE];
#pragma unroll
  for (int s = 0; s < DSTATE; ++s) h[s] = 0.0f;

  __syncthreads();

  const int ntiles = L / TT;
  for (int t = 0; t < ntiles; ++t) {
    const int l0 = t * TT;

    // ---- S0: gather x tile (16 tokens x 64 ch), f32 -> bf16 --------------
    for (int e = tid; e < TT * 64; e += 128) {
      int tok = e >> 6, c = e & 63;
      tA[tok * 64 + c] = (bf16)x[base0 + (long)(l0 + tok) * lstride + (long)c * CH_STRIDE];
    }
    __syncthreads();

    // ---- S1: xz = tileA(16x64) @ W_in^T(64x256), WMMA -------------------
    for (int i = 0; i < 4; ++i) {
      int nt = wid * 4 + i;                       // 16 n-tiles over 4 waves
      v8f acc = {};
#pragma unroll
      for (int kf = 0; kf < 2; ++kf) {
        v16bf a = frag_ld(&tA[row * 64 + kf * 32], half);
        v16bf b = frag_ld(&sWin[(nt * 16 + row) * 64 + kf * 32], half);
        acc = wmma_bf16(a, b, acc);
      }
      int col = nt * 16 + row;
#pragma unroll
      for (int r = 0; r < 8; ++r) sxz[(r + 8 * half) * 256 + col] = acc[r];
    }
    __syncthreads();

    // ---- S2: depthwise causal conv(4) + bias + SiLU ----------------------
    for (int e = tid; e < TT * 128; e += 128) {
      int tok = e >> 7, d = e & 127;
      float acc = scb[d];
#pragma unroll
      for (int j = 0; j < 4; ++j) {
        int li = tok - 3 + j;
        float v = (li >= 0) ? sxz[li * 256 + d] : shist[(li + 3) * 128 + d];
        acc += scw[d * 4 + j] * v;
      }
      float xcv = siluf_(acc);
      sxc[tok * 128 + d]  = xcv;
      sxcb[tok * 128 + d] = (bf16)xcv;
    }
    __syncthreads();

    // ---- S3: GEMM2 dbl = xc(16x128) @ W_x^T (waves 0-2); wave 3 rolls history
    if (wid < 3) {
      int nn = wid * 16 + row;                    // n-tile = wid (N padded to 48)
      v8f acc = {};
#pragma unroll
      for (int kf = 0; kf < 4; ++kf) {
        v16bf a = frag_ld(&sxcb[row * 128 + kf * 32], half);
        v16bf b = frag_ld(&sWx[nn * 128 + kf * 32], half);
        acc = wmma_bf16(a, b, acc);
      }
#pragma unroll
      for (int r = 0; r < 8; ++r) sdbl[(r + 8 * half) * 48 + nn] = acc[r];
    } else {
      for (int e = lane; e < 3 * 128; e += 32) {  // save raw tokens 13..15
        int rrow = e >> 7, d = e & 127;
        shist[rrow * 128 + d] = sxz[(13 + rrow) * 256 + d];
      }
    }
    __syncthreads();

    // ---- S4: dt = softplus(dt_raw @ W_dt^T + b_dt) -----------------------
    for (int e = tid; e < TT * 128; e += 128) {
      int tok = e >> 7, d = e & 127;
      float s = sbdt[d];
#pragma unroll
      for (int r = 0; r < DTRANK; ++r) s += sdbl[tok * 48 + r] * sWdt[d * 4 + r];
      sdt[tok * 128 + d] = softplusf_(s);
    }
    __syncthreads();

    // ---- S5: selective scan (serial in tokens, parallel over 128 channels)
    {
      const int d = tid;                          // one D_INNER channel / thread
      for (int tok = 0; tok < TT; ++tok) {
        float dtv = sdt[tok * 128 + d];
        float xv  = sxc[tok * 128 + d];
        float dx  = dtv * xv;
        float y   = 0.0f;
#pragma unroll
        for (int s = 0; s < DSTATE; ++s) {
          float dA = __expf(dtv * sA[d * 16 + s]);
          h[s] = h[s] * dA + dx * sdbl[tok * 48 + 4 + s];   // B_t
          y   += h[s] * sdbl[tok * 48 + 20 + s];            // C_t
        }
        float zv = sxz[tok * 256 + 128 + d];
        float yo = (y + xv * sDp[d]) * siluf_(zv);
        syb[tok * 128 + d] = (bf16)yo;
      }
    }
    __syncthreads();

    // ---- S6: out = y(16x128) @ W_out^T(128x64), WMMA, scatter to global --
    {
      int c = wid * 16 + row;                     // 4 n-tiles over 4 waves
      v8f acc = {};
#pragma unroll
      for (int kf = 0; kf < 4; ++kf) {
        v16bf a = frag_ld(&syb[row * 128 + kf * 32], half);
        v16bf b = frag_ld(&sWout[c * 128 + kf * 32], half);
        acc = wmma_bf16(a, b, acc);
      }
#pragma unroll
      for (int r = 0; r < 8; ++r) {
        int m = r + 8 * half;
        out[base0 + (long)(l0 + m) * lstride + (long)c * CH_STRIDE] = acc[r];
      }
    }
    __syncthreads();                              // protect tA/sxz/syb reuse
  }
}

// ---------------------------------------------------------------------------
// Fuse: f = gelu(cat[sp;se](NPOS x 128) @ W_fuse^T + b_fuse); accumulate
// per-channel sum/sumsq for the (0,2,3,4)-axis batch statistics.
// ---------------------------------------------------------------------------
__global__ __launch_bounds__(128)
void smb_fuse_kernel(const float* __restrict__ sp, const float* __restrict__ se,
                     const float* __restrict__ W_fuse, const float* __restrict__ b_fuse,
                     float* __restrict__ fbuf, float* __restrict__ stats)
{
  __shared__ bf16  sWf [64 * 128];
  __shared__ bf16  scat[16 * 128];
  __shared__ float psum[64], psq[64];

  const int tid  = threadIdx.x;
  const int lane = tid & 31;
  const int half = (lane >> 4) & 1;
  const int wid  = tid >> 5;
  const int row  = lane & 15;
  const int p0   = blockIdx.x * 16;               // token tile

  for (int e = tid; e < 64 * 128; e += 128) sWf[e] = (bf16)W_fuse[e];
  if (tid < 64) { psum[tid] = 0.0f; psq[tid] = 0.0f; }
  for (int e = tid; e < 16 * 128; e += 128) {
    int tok = e >> 7, c = e & 127;
    int p = p0 + tok, b = p >> 15, rem = p & 32767;
    const float* src = (c < 64) ? sp : se;
    scat[e] = (bf16)src[(long)b * B_STRIDE + (long)(c & 63) * CH_STRIDE + rem];
  }
  __syncthreads();

  int nn = wid * 16 + row;                        // output channel
  v8f acc = {};
#pragma unroll
  for (int kf = 0; kf < 4; ++kf) {
    v16bf a = frag_ld(&scat[row * 128 + kf * 32], half);
    v16bf b = frag_ld(&sWf[nn * 128 + kf * 32], half);
    acc = wmma_bf16(a, b, acc);
  }
  float bias = b_fuse[nn];
  float ls = 0.0f, lsq = 0.0f;
#pragma unroll
  for (int r = 0; r < 8; ++r) {
    int m = r + 8 * half;
    int p = p0 + m, b = p >> 15, rem = p & 32767;
    float g = geluf_(acc[r] + bias);
    fbuf[(long)b * B_STRIDE + (long)nn * CH_STRIDE + rem] = g;
    ls += g; lsq += g * g;
  }
  atomicAdd(&psum[nn], ls);
  atomicAdd(&psq[nn], lsq);
  __syncthreads();
  if (tid < 64) {
    atomicAdd(&stats[tid],      psum[tid]);
    atomicAdd(&stats[64 + tid], psq[tid]);
  }
}

__global__ void smb_init_stats(float* __restrict__ stats)
{
  if (threadIdx.x < 128) stats[threadIdx.x] = 0.0f;
}

__global__ void smb_norm_kernel(const float* __restrict__ fbuf, const float* __restrict__ stats,
                                const float* __restrict__ gamma, const float* __restrict__ beta,
                                const float* __restrict__ x, float* __restrict__ out, int ntot)
{
  int idx = blockIdx.x * blockDim.x + threadIdx.x;
  if (idx >= ntot) return;
  int o = (idx >> 15) & 63;                       // channel in (B,64,H,W,DD)
  const float invN = 1.0f / (float)NPOS;
  float mean = stats[o] * invN;
  float var  = stats[64 + o] * invN - mean * mean;
  float istd = rsqrtf(var + 1e-5f);
  out[idx] = (fbuf[idx] - mean) * istd * gamma[o] + beta[o] + x[idx];
}

// ---------------------------------------------------------------------------
extern "C" void kernel_launch(void* const* d_in, const int* in_sizes, int n_in,
                              void* d_out, int out_size, void* d_ws, size_t ws_size,
                              hipStream_t stream)
{
  (void)in_sizes; (void)n_in; (void)out_size; (void)ws_size;
  // setup_inputs() dict order:
  const float* spa_W_in   = (const float*)d_in[0];
  const float* spa_conv_w = (const float*)d_in[1];
  const float* spa_conv_b = (const float*)d_in[2];
  const float* spa_W_x    = (const float*)d_in[3];
  const float* spa_W_dt   = (const float*)d_in[4];
  const float* spa_b_dt   = (const float*)d_in[5];
  const float* spa_A_log  = (const float*)d_in[6];
  const float* spa_Dp     = (const float*)d_in[7];
  const float* spa_W_out  = (const float*)d_in[8];
  const float* spe_W_in   = (const float*)d_in[9];
  const float* spe_conv_w = (const float*)d_in[10];
  const float* spe_conv_b = (const float*)d_in[11];
  const float* spe_W_x    = (const float*)d_in[12];
  const float* spe_W_dt   = (const float*)d_in[13];
  const float* spe_b_dt   = (const float*)d_in[14];
  const float* spe_A_log  = (const float*)d_in[15];
  const float* spe_Dp     = (const float*)d_in[16];
  const float* spe_W_out  = (const float*)d_in[17];
  const float* W_fuse     = (const float*)d_in[18];
  const float* b_fuse     = (const float*)d_in[19];
  const float* gamma      = (const float*)d_in[20];
  const float* beta       = (const float*)d_in[21];
  const float* x          = (const float*)d_in[22];
  float* out = (float*)d_out;

  const int TOT = NB * 64 * HH * WW * DDp;        // 4,194,304
  float* ws    = (float*)d_ws;
  float* sp    = ws;
  float* se    = ws + (size_t)TOT;
  float* fb    = ws + (size_t)2 * TOT;
  float* stats = ws + (size_t)3 * TOT;

  // spatial: 64 sequences of L=1024 ; spectral: 2048 sequences of L=32
  smb_mamba_kernel<<<64, 128, 0, stream>>>(x, spa_W_in, spa_conv_w, spa_conv_b,
      spa_W_x, spa_W_dt, spa_b_dt, spa_A_log, spa_Dp, spa_W_out, sp, 0, 1024);
  smb_mamba_kernel<<<2048, 128, 0, stream>>>(x, spe_W_in, spe_conv_w, spe_conv_b,
      spe_W_x, spe_W_dt, spe_b_dt, spe_A_log, spe_Dp, spe_W_out, se, 1, 32);
  smb_init_stats<<<1, 128, 0, stream>>>(stats);
  smb_fuse_kernel<<<NPOS / 16, 128, 0, stream>>>(sp, se, W_fuse, b_fuse, fb, stats);
  smb_norm_kernel<<<(TOT + 255) / 256, 256, 0, stream>>>(fb, stats, gamma, beta, x, out, TOT);
}
